// UniPhyBlock_72773925863915
// MI455X (gfx1250) — matrix operand order repaired and checked by reading, not compiled, not run on hardware
//
#include <hip/hip_runtime.h>
#include <hip/hip_bf16.h>

// ---------------------------------------------------------------------------
// UniPhyBlock for MI455X (gfx1250): wave32 + WMMA f16 (fp32 accum) pipeline.
//   B=1 T=16 D=64 H=W=48 -> HW=2304, NPIX=36864, C=2D=128, F=128, NE=8
// All WMMA B operands (and the conv A) are staged in LDS in *fragment-swizzled* layout so each
// lane loads its 16 halves with contiguous b128 pairs (no scalar ds gathers).
// ---------------------------------------------------------------------------

#define BT   16
#define DIM  64
#define C2   128
#define HH   48
#define WW   48
#define HW   2304
#define NPIX (BT * HW)      // 36864
#define FF   128
#define KCONV 1152

typedef __attribute__((ext_vector_type(16))) _Float16 v16h;
typedef __attribute__((ext_vector_type(8)))  _Float16 v8h;
typedef __attribute__((ext_vector_type(8)))  float    v8f;

#define WMMA_F16(A, Bm, Cacc) \
  __builtin_amdgcn_wmma_f32_16x16x32_f16(false, (A), false, (Bm), (short)0, (Cacc), false, false)

// Fragment-swizzled storage: one 32x16 B-fragment (or 16x32 A-fragment) is 512
// halves, lane-major: lane L's 16 elements are contiguous at frag + L*16.
__device__ __forceinline__ v16h frag_ld(const _Float16* frag, int lane) {
  return *(const v16h*)(frag + (lane << 4));
}

// B element (kin 0..31, nlo 0..15) -> offset inside a 512-half fragment.
// ISA B layout: lane = ((kin>>3)&1)*16 + nlo ; element = (kin & 7) + 8*(kin>>4)
__device__ __forceinline__ int bswz(int kin, int nlo) {
  return (((((kin >> 3) & 1) << 4) + nlo) << 4) + (kin & 7) + ((kin >> 4) << 3);
}

// A fragment gathered from row-major global memory (leading dim ld):
// lane L: row = L&15, K group = (L>>4)*8 ; elems 0..7 -> K=kk+e, 8..15 -> 16+kk+e
__device__ __forceinline__ v16h frag_a_g(const _Float16* base, int ld, int lane) {
  const _Float16* p = base + (lane & 15) * ld + ((lane >> 4) << 3);
  v8h lo = *(const v8h*)p;
  v8h hi = *(const v8h*)(p + 16);
  return __builtin_shufflevector(lo, hi, 0, 1, 2, 3, 4, 5, 6, 7,
                                 8, 9, 10, 11, 12, 13, 14, 15);
}

__device__ __forceinline__ v16h neg16(v16h x) {
  v16h r;
#pragma unroll
  for (int e = 0; e < 16; ++e) r[e] = -x[e];
  return r;
}

__device__ __forceinline__ float gelu_tanh(float x) {
  float x3 = x * x * x;
  return 0.5f * x * (1.f + tanhf(0.7978845608f * (x + 0.044715f * x3)));
}

// ------------------------------------------------------------------- pack
// conv weights + E matrices -> half, already in fragment-swizzled order.
__global__ void k_pack(const float* __restrict__ w_conv,
                       const float* __restrict__ eer, const float* __restrict__ eei,
                       const float* __restrict__ edr, const float* __restrict__ edi,
                       _Float16* __restrict__ wb,
                       _Float16* __restrict__ ebr, _Float16* __restrict__ ebi,
                       _Float16* __restrict__ dbr, _Float16* __restrict__ dbi) {
  int i = blockIdx.x * 256 + threadIdx.x;
  if (i < KCONV * C2) {
    int k = i >> 7, n = i & 127;     // GEMM K = (kh*3+kw)*128 + ci
    int khkw = k >> 7, ci = k & 127;
    float v = w_conv[((size_t)(n * 128 + ci)) * 9 + khkw];
    int d = (((k >> 5) * 8 + (n >> 4)) << 9) + bswz(k & 31, n & 15);
    wb[d] = (_Float16)v;
  }
  if (i < DIM * DIM) {
    int k = i >> 6, n = i & 63;
    int d = (((k >> 5) * 4 + (n >> 4)) << 9) + bswz(k & 31, n & 15);
    ebr[d] = (_Float16)eer[i]; ebi[d] = (_Float16)eei[i];
    dbr[d] = (_Float16)edr[i]; dbi[d] = (_Float16)edi[i];
  }
}

// ------------------------------------------------------------- spatial LN
__global__ void k_ln_spatial(const float* __restrict__ xre, const float* __restrict__ xim,
                             const float* __restrict__ gamma, const float* __restrict__ beta,
                             _Float16* __restrict__ xn) {
  int wave = threadIdx.x >> 5, lane = threadIdx.x & 31;
  int pix = blockIdx.x * 8 + wave;                 // bt*HW + hw
  int bt = pix / HW, hw = pix - bt * HW;
  float v[4];
#pragma unroll
  for (int j = 0; j < 4; ++j) {
    int c = lane + 32 * j;
    const float* src = (c < DIM) ? xre : xim;
    int d = c & 63;
    v[j] = src[((size_t)(bt * DIM + d)) * HW + hw];
  }
  float s = v[0] + v[1] + v[2] + v[3];
  float q = v[0] * v[0] + v[1] * v[1] + v[2] * v[2] + v[3] * v[3];
#pragma unroll
  for (int off = 16; off > 0; off >>= 1) { s += __shfl_xor(s, off); q += __shfl_xor(q, off); }
  float m = s * (1.f / 128.f);
  float r = rsqrtf(q * (1.f / 128.f) - m * m + 1e-5f);
#pragma unroll
  for (int j = 0; j < 4; ++j) {
    int c = lane + 32 * j;
    xn[(size_t)pix * C2 + c] = (_Float16)((v[j] - m) * r * gamma[c] + beta[c]);
  }
}

// -------------------------------------------------- conv3x3 implicit GEMM
// block: 64 pixels x 128 cout ; K = 9 taps x 4 cin-chunks of 32.
// 36 blocks per image (2304/64) -> a block never straddles images, so all
// pixel geometry is hoisted out of the K loop (no divisions in the hot loop).
__global__ void k_conv(const _Float16* __restrict__ xn, const _Float16* __restrict__ wb,
                       const float* __restrict__ b_conv, const float* __restrict__ metric,
                       const float* __restrict__ x_re, const float* __restrict__ x_im,
                       _Float16* __restrict__ xe_rh, _Float16* __restrict__ xe_ih) {
  __shared__ __align__(32) _Float16 As[4 * 512];   // 4 m-strips, frag-swizzled
  __shared__ __align__(32) _Float16 Bs[8 * 512];   // 8 n-tiles, frag-swizzled
  int lane = threadIdx.x & 31, wave = threadIdx.x >> 5;
  int m0 = 16 * (wave >> 1), n0w = (wave & 1) * 4;   // n-tile base index
  int btBlk = blockIdx.x / 36;
  int hwBase = (blockIdx.x - btBlk * 36) * 64;

  // per-thread A-staging constants (loop-invariant)
  int row = threadIdx.x >> 2, c8 = (threadIdx.x & 3) * 8;
  int hwA = hwBase + row;
  int yy = hwA / WW, xx = hwA - yy * WW;
  const _Float16* aBase =
      xn + ((size_t)btBlk * HW + (size_t)yy * WW + xx) * C2 + c8;
  // A swizzle: lane = (m&15) + 16*((c8>>3)&1) ; e0 = 8*(c8>>4) (contiguous)
  int aDst = ((row >> 4) << 9) + (((row & 15) + (((c8 >> 3) & 1) << 4)) << 4) +
             ((c8 >> 4) << 3);
  const _Float16* aFrag = As + ((m0 >> 4) << 9);

  v8f acc[4] = {};
  int kc = 0;
  for (int khkw = 0; khkw < 9; ++khkw) {
    int dy = khkw / 3 - 1, dx = khkw - (khkw / 3) * 3 - 1;
    bool inb = ((unsigned)(yy + dy) < HH) & ((unsigned)(xx + dx) < WW);
    const _Float16* aSrc = aBase + (dy * WW + dx) * C2;
#pragma unroll
    for (int c4 = 0; c4 < 4; ++c4, ++kc) {
      if (kc < 35)  // gfx1250 global_prefetch path for next filter chunk
        __builtin_prefetch(wb + (size_t)(kc + 1) * 4096, 0, 1);
      v8h av = {};
      if (inb) av = *(const v8h*)(aSrc + c4 * 32);
      *(v8h*)(As + aDst) = av;
      {  // B tile: straight contiguous copy (already swizzled in global)
        const v8h* s = (const v8h*)(wb + (size_t)kc * 4096);
        v8h* d = (v8h*)Bs;
        d[threadIdx.x] = s[threadIdx.x];
        d[threadIdx.x + 256] = s[threadIdx.x + 256];
      }
      __syncthreads();
      v16h a = frag_ld(aFrag, lane);
#pragma unroll
      for (int nt = 0; nt < 4; ++nt) {
        v16h b = frag_ld(Bs + ((n0w + nt) << 9), lane);
        acc[nt] = WMMA_F16(a, b, acc[nt]);
      }
      __syncthreads();
    }
  }
  // epilogue: + b_conv, * (1+metric), + x residual, split complex, store half
  int nlo = lane & 15, hs = lane >> 4;
#pragma unroll
  for (int nt = 0; nt < 4; ++nt) {
    int col = (n0w + nt) * 16 + nlo;
    float bc = b_conv[col];
#pragma unroll
    for (int r = 0; r < 8; ++r) {
      int hw = hwBase + m0 + r + 8 * hs;
      size_t p = (size_t)btBlk * HW + hw;
      float val = (acc[nt][r] + bc) * (1.f + metric[hw]);
      if (col < DIM) {
        float xv = x_re[((size_t)(btBlk * DIM + col)) * HW + hw];
        xe_rh[p * DIM + col] = (_Float16)(xv + val);
      } else {
        int d2 = col - DIM;
        float xv = x_im[((size_t)(btBlk * DIM + d2)) * HW + hw];
        xe_ih[p * DIM + d2] = (_Float16)(xv + val);
      }
    }
  }
}

// ------------------------------------------ x_eigen = xe @ E_enc (complex)
__global__ void k_eigen(const _Float16* __restrict__ xe_rh, const _Float16* __restrict__ xe_ih,
                        const _Float16* __restrict__ Er, const _Float16* __restrict__ Ei,
                        float* __restrict__ xeig_r, float* __restrict__ xeig_i) {
  __shared__ __align__(32) _Float16 Br[8 * 512];   // 2 kchunks x 4 ntiles
  __shared__ __align__(32) _Float16 Bi[8 * 512];
  {
    const v8h* sr = (const v8h*)Er;
    const v8h* si = (const v8h*)Ei;
    v8h* dr = (v8h*)Br;
    v8h* di = (v8h*)Bi;
    for (int i = threadIdx.x; i < 512; i += 256) { dr[i] = sr[i]; di[i] = si[i]; }
  }
  __syncthreads();
  int lane = threadIdx.x & 31, wave = threadIdx.x >> 5;
  int row0 = blockIdx.x * 128 + wave * 16;
  v8f accR[4] = {}, accI[4] = {};
#pragma unroll
  for (int kc = 0; kc < 2; ++kc) {
    int k0 = kc * 32;
    v16h ar = frag_a_g(xe_rh + (size_t)row0 * DIM + k0, DIM, lane);
    v16h ai = frag_a_g(xe_ih + (size_t)row0 * DIM + k0, DIM, lane);
    v16h nai = neg16(ai);
#pragma unroll
    for (int nt = 0; nt < 4; ++nt) {
      v16h br = frag_ld(Br + ((kc * 4 + nt) << 9), lane);
      v16h bi = frag_ld(Bi + ((kc * 4 + nt) << 9), lane);
      accR[nt] = WMMA_F16(ar, br, accR[nt]);
      accR[nt] = WMMA_F16(nai, bi, accR[nt]);
      accI[nt] = WMMA_F16(ar, bi, accI[nt]);
      accI[nt] = WMMA_F16(ai, br, accI[nt]);
    }
  }
  int nlo = lane & 15, hs = lane >> 4;
#pragma unroll
  for (int nt = 0; nt < 4; ++nt)
#pragma unroll
    for (int r = 0; r < 8; ++r) {
      size_t p = (size_t)row0 + r + 8 * hs;
      size_t ix = p * DIM + nt * 16 + nlo;
      xeig_r[ix] = accR[nt][r];
      xeig_i[ix] = accI[nt][r];
    }
}

// ---------------------------------------------------------- x_mean (bt,d)
__global__ void k_xmean(const float* __restrict__ xeig_r, const float* __restrict__ xeig_i,
                        float* __restrict__ xm_r, float* __restrict__ xm_i) {
  __shared__ float sr[256], si[256];
  int bt = blockIdx.x >> 6, d = blockIdx.x & 63;
  float ar = 0.f, ai = 0.f;
  for (int hw = threadIdx.x; hw < HW; hw += 256) {
    size_t ix = ((size_t)bt * HW + hw) * DIM + d;
    ar += xeig_r[ix]; ai += xeig_i[ix];
  }
  sr[threadIdx.x] = ar; si[threadIdx.x] = ai;
  __syncthreads();
  for (int o = 128; o > 0; o >>= 1) {
    if (threadIdx.x < o) { sr[threadIdx.x] += sr[threadIdx.x + o]; si[threadIdx.x] += si[threadIdx.x + o]; }
    __syncthreads();
  }
  if (threadIdx.x == 0) { xm_r[blockIdx.x] = sr[0] * (1.f / HW); xm_i[blockIdx.x] = si[0] * (1.f / HW); }
}

// ------------------------------------- flux scan + gate + decay operators
__global__ void k_flux(const float* __restrict__ xm_r, const float* __restrict__ xm_i,
                       const float* __restrict__ f0r, const float* __restrict__ f0i,
                       const float* __restrict__ a_re, const float* __restrict__ a_im,
                       const float* __restrict__ wsr, const float* __restrict__ wsi,
                       const float* __restrict__ w_gate, const float* __restrict__ b_gate,
                       const float* __restrict__ lam_raw, const float* __restrict__ lam_im,
                       const float* __restrict__ dt,
                       float* __restrict__ gate_g, float* __restrict__ src_r, float* __restrict__ src_i,
                       float* __restrict__ A_r, float* __restrict__ A_i,
                       float* __restrict__ opf_r, float* __restrict__ opf_i) {
  __shared__ float fr[DIM], fi[DIM];
  int d = threadIdx.x;  // 64 threads
  float ar = a_re[d], ai = a_im[d];
  float cr = f0r[d], ci = f0i[d];
  float lraw = lam_raw[d];
  float sp = (lraw > 20.f) ? lraw : log1pf(expf(lraw));
  float lr = -sp, li = lam_im[d];
  float inv_l2 = 1.f / (lr * lr + li * li);
  for (int t = 0; t < BT; ++t) {
    float xr = xm_r[t * DIM + d], xi = xm_i[t * DIM + d];
    float nr = ar * cr - ai * ci + (1.f - ar) * xr + ai * xi;
    float ni = ar * ci + ai * cr + (1.f - ar) * xi - ai * xr;
    cr = nr; ci = ni;
    fr[d] = cr; fi[d] = ci;
    __syncthreads();
    float sre = 0.f, sim = 0.f, g = b_gate[d];
    for (int k = 0; k < DIM; ++k) {
      float wr = wsr[k * DIM + d], wi = wsi[k * DIM + d];
      sre += fr[k] * wr - fi[k] * wi;
      sim += fr[k] * wi + fi[k] * wr;
      g += fr[k] * w_gate[k * DIM + d] + fi[k] * w_gate[(k + DIM) * DIM + d];
    }
    g = 1.f / (1.f + expf(-g));
    int te = t * DIM + d;
    gate_g[te] = g; src_r[te] = sre; src_i[te] = sim;
    float dte = dt[t];
    float er = expf(lr * dte);
    float Are = er * cosf(li * dte), Aim = er * sinf(li * dte);
    A_r[te] = Are; A_i[te] = Aim;
    float numr = Are - 1.f, numi = Aim;
    opf_r[te] = (numr * lr + numi * li) * inv_l2;
    opf_i[te] = (numi * lr - numr * li) * inv_l2;
    __syncthreads();
  }
}

// ---------------------------------------- temporal scan: u_t = A u_{t-1}+Xs
__global__ void k_tscan(const float* __restrict__ xeig_r, const float* __restrict__ xeig_i,
                        const float* __restrict__ gate_g, const float* __restrict__ src_r,
                        const float* __restrict__ src_i,
                        const float* __restrict__ A_r, const float* __restrict__ A_i,
                        const float* __restrict__ opf_r, const float* __restrict__ opf_i,
                        _Float16* __restrict__ u_rh, _Float16* __restrict__ u_ih) {
  int tid = blockIdx.x * 256 + threadIdx.x;
  int e = tid & 63, hw = tid >> 6;
  float ur = 0.f, ui = 0.f;
#pragma unroll
  for (int t = 0; t < BT; ++t) {
    size_t ix = ((size_t)t * HW + hw) * DIM + e;
    int te = t * DIM + e;
    float g = gate_g[te];
    float fre = xeig_r[ix] * g + src_r[te] * (1.f - g);
    float fim = xeig_i[ix] * g + src_i[te] * (1.f - g);
    float pr = opf_r[te], pi = opf_i[te];
    float Xr = fre * pr - fim * pi, Xi = fre * pi + fim * pr;
    float Ar = A_r[te], Ai = A_i[te];
    float nr = Ar * ur - Ai * ui + Xr;
    float ni = Ar * ui + Ai * ur + Xi;
    ur = nr; ui = ni;
    u_rh[ix] = (_Float16)ur; u_ih[ix] = (_Float16)ui;
  }
}

// ----------------------- x_dec = u @ E_dec (complex WMMA) fused temporal LN
__global__ void k_dec(const _Float16* __restrict__ u_rh, const _Float16* __restrict__ u_ih,
                      const _Float16* __restrict__ Edr, const _Float16* __restrict__ Edi,
                      const float* __restrict__ gamma_t, const float* __restrict__ beta_t,
                      float* __restrict__ xoc_r, float* __restrict__ xoc_i,
                      _Float16* __restrict__ xoc_rh, _Float16* __restrict__ xoc_ih) {
  __shared__ __align__(32) _Float16 Br[8 * 512];
  __shared__ __align__(32) _Float16 Bi[8 * 512];
  {
    const v8h* sr = (const v8h*)Edr;
    const v8h* si = (const v8h*)Edi;
    v8h* dr = (v8h*)Br;
    v8h* di = (v8h*)Bi;
    for (int i = threadIdx.x; i < 512; i += 256) { dr[i] = sr[i]; di[i] = si[i]; }
  }
  __syncthreads();
  int lane = threadIdx.x & 31, wave = threadIdx.x >> 5;
  int row0 = blockIdx.x * 128 + wave * 16;
  v8f accR[4] = {}, accI[4] = {};
#pragma unroll
  for (int kc = 0; kc < 2; ++kc) {
    int k0 = kc * 32;
    v16h ar = frag_a_g(u_rh + (size_t)row0 * DIM + k0, DIM, lane);
    v16h ai = frag_a_g(u_ih + (size_t)row0 * DIM + k0, DIM, lane);
    v16h nai = neg16(ai);
#pragma unroll
    for (int nt = 0; nt < 4; ++nt) {
      v16h br = frag_ld(Br + ((kc * 4 + nt) << 9), lane);
      v16h bi = frag_ld(Bi + ((kc * 4 + nt) << 9), lane);
      accR[nt] = WMMA_F16(ar, br, accR[nt]);
      accR[nt] = WMMA_F16(nai, bi, accR[nt]);
      accI[nt] = WMMA_F16(ar, bi, accI[nt]);
      accI[nt] = WMMA_F16(ai, br, accI[nt]);
    }
  }
  // LayerNorm over 128 = concat(re, im) per output row, in-wave.
  float mean[8], rstd[8];
#pragma unroll
  for (int r = 0; r < 8; ++r) {
    float s = 0.f, q = 0.f;
#pragma unroll
    for (int nt = 0; nt < 4; ++nt) {
      float a = accR[nt][r], b = accI[nt][r];
      s += a + b; q += a * a + b * b;
    }
#pragma unroll
    for (int off = 1; off < 16; off <<= 1) { s += __shfl_xor(s, off); q += __shfl_xor(q, off); }
    float m = s * (1.f / 128.f);
    mean[r] = m;
    rstd[r] = rsqrtf(q * (1.f / 128.f) - m * m + 1e-5f);
  }
  int nlo = lane & 15, hs = lane >> 4;
#pragma unroll
  for (int nt = 0; nt < 4; ++nt) {
    int d = nt * 16 + nlo;
    float gr = gamma_t[d], br0 = beta_t[d];
    float gi = gamma_t[d + DIM], bi0 = beta_t[d + DIM];
#pragma unroll
    for (int r = 0; r < 8; ++r) {
      size_t p = (size_t)row0 + r + 8 * hs;
      float vr = (accR[nt][r] - mean[r]) * rstd[r] * gr + br0;
      float vi = (accI[nt][r] - mean[r]) * rstd[r] * gi + bi0;
      size_t ix = p * DIM + d;
      xoc_r[ix] = vr; xoc_i[ix] = vi;
      xoc_rh[ix] = (_Float16)vr; xoc_ih[ix] = (_Float16)vi;
    }
  }
}

// ---------------------------------------------------------------- feats
__global__ void k_feats(const float* __restrict__ xoc_r, const float* __restrict__ xoc_i,
                        float* __restrict__ feats) {
  __shared__ float sm[256];
  int c = blockIdx.x & 127, t = blockIdx.x >> 7;
  const float* src = (c < DIM) ? xoc_r : xoc_i;
  int d = c & 63;
  float s = 0.f;
  for (int hw = threadIdx.x; hw < HW; hw += 256)
    s += src[((size_t)t * HW + hw) * DIM + d];
  sm[threadIdx.x] = s;
  __syncthreads();
  for (int o = 128; o > 0; o >>= 1) {
    if (threadIdx.x < o) sm[threadIdx.x] += sm[threadIdx.x + o];
    __syncthreads();
  }
  if (threadIdx.x == 0) feats[t * 128 + c] = sm[0] * (1.f / HW);
}

// --------------------------------------------------------------- router
__global__ void k_router(const float* __restrict__ feats, const float* __restrict__ w_r,
                         const float* __restrict__ b_r, int* __restrict__ idx_d,
                         float* __restrict__ ptop_d) {
  __shared__ float lg[BT][8];
  int t = threadIdx.x >> 3, j = threadIdx.x & 7;   // 128 threads
  float acc = b_r[j];
  for (int c = 0; c < 128; ++c) acc += feats[t * 128 + c] * w_r[c * 8 + j];
  lg[t][j] = acc;
  __syncthreads();
  if (j == 0) {
    float m = lg[t][0]; int bi = 0;
    for (int k = 1; k < 8; ++k)
      if (lg[t][k] > m) { m = lg[t][k]; bi = k; }
    float den = 0.f;
    for (int k = 0; k < 8; ++k) den += expf(lg[t][k] - m);
    idx_d[t] = bi;
    ptop_d[t] = 1.f / den;
  }
}

// --------------------------- MoE layer 1: hdn = gelu_c(xm @ W1[idx] + b1)
__global__ void k_moe1(const _Float16* __restrict__ xoc_rh, const _Float16* __restrict__ xoc_ih,
                       const float* __restrict__ w1_re, const float* __restrict__ w1_im,
                       const float* __restrict__ b1_re, const float* __restrict__ b1_im,
                       const int* __restrict__ idx_d,
                       _Float16* __restrict__ hdn_rh, _Float16* __restrict__ hdn_ih) {
  __shared__ __align__(32) _Float16 Br[16 * 512];   // 2 kchunks x 8 ntiles
  __shared__ __align__(32) _Float16 Bi[16 * 512];
  int timg = blockIdx.y;
  int expert = idx_d[timg];
  const float* w1r = w1_re + (size_t)expert * DIM * FF;
  const float* w1i = w1_im + (size_t)expert * DIM * FF;
  for (int i = threadIdx.x; i < DIM * FF; i += 256) {   // swizzle once
    int k = i >> 7, n = i & 127;
    int d = (((k >> 5) * 8 + (n >> 4)) << 9) + bswz(k & 31, n & 15);
    Br[d] = (_Float16)w1r[i];
    Bi[d] = (_Float16)w1i[i];
  }
  __syncthreads();
  int lane = threadIdx.x & 31, wave = threadIdx.x >> 5;
  int m0 = 16 * (wave >> 1), n0w = (wave & 1) * 4;
  size_t rowBase = (size_t)timg * HW + (size_t)blockIdx.x * 64;
  v8f accR[4] = {}, accI[4] = {};
#pragma unroll
  for (int kc = 0; kc < 2; ++kc) {
    int k0 = kc * 32;
    v16h ar = frag_a_g(xoc_rh + (rowBase + m0) * DIM + k0, DIM, lane);
    v16h ai = frag_a_g(xoc_ih + (rowBase + m0) * DIM + k0, DIM, lane);
    v16h nai = neg16(ai);
#pragma unroll
    for (int nt = 0; nt < 4; ++nt) {
      v16h br = frag_ld(Br + ((kc * 8 + n0w + nt) << 9), lane);
      v16h bi = frag_ld(Bi + ((kc * 8 + n0w + nt) << 9), lane);
      accR[nt] = WMMA_F16(ar, br, accR[nt]);
      accR[nt] = WMMA_F16(nai, bi, accR[nt]);
      accI[nt] = WMMA_F16(ar, bi, accI[nt]);
      accI[nt] = WMMA_F16(ai, br, accI[nt]);
    }
  }
  const float* b1r = b1_re + expert * FF;
  const float* b1i = b1_im + expert * FF;
  int nlo = lane & 15, hs = lane >> 4;
#pragma unroll
  for (int nt = 0; nt < 4; ++nt) {
    int col = (n0w + nt) * 16 + nlo;
    float br0 = b1r[col], bi0 = b1i[col];
#pragma unroll
    for (int r = 0; r < 8; ++r) {
      size_t row = rowBase + m0 + r + 8 * hs;
      float zr = gelu_tanh(accR[nt][r] + br0);
      float zi = gelu_tanh(accI[nt][r] + bi0);
      hdn_rh[row * FF + col] = (_Float16)zr;
      hdn_ih[row * FF + col] = (_Float16)zi;
    }
  }
}

// ------------- MoE layer 2 + ptop*delta + residuals + final output store
__global__ void k_moe2(const _Float16* __restrict__ hdn_rh, const _Float16* __restrict__ hdn_ih,
                       const float* __restrict__ w2_re, const float* __restrict__ w2_im,
                       const float* __restrict__ b2_re, const float* __restrict__ b2_im,
                       const int* __restrict__ idx_d, const float* __restrict__ ptop_d,
                       const float* __restrict__ xoc_r, const float* __restrict__ xoc_i,
                       const float* __restrict__ x_re, const float* __restrict__ x_im,
                       float* __restrict__ out) {
  __shared__ __align__(32) _Float16 Br[16 * 512];   // 4 kchunks x 4 ntiles
  __shared__ __align__(32) _Float16 Bi[16 * 512];
  int timg = blockIdx.y;
  int expert = idx_d[timg];
  const float* w2r = w2_re + (size_t)expert * FF * DIM;
  const float* w2i = w2_im + (size_t)expert * FF * DIM;
  for (int i = threadIdx.x; i < FF * DIM; i += 256) {   // swizzle once
    int k = i >> 6, n = i & 63;
    int d = (((k >> 5) * 4 + (n >> 4)) << 9) + bswz(k & 31, n & 15);
    Br[d] = (_Float16)w2r[i];
    Bi[d] = (_Float16)w2i[i];
  }
  __syncthreads();
  int lane = threadIdx.x & 31, wave = threadIdx.x >> 5;
  int m0 = 16 * (wave & 3), n0w = (wave >> 2) * 2;
  size_t rowBase = (size_t)timg * HW + (size_t)blockIdx.x * 64;
  v8f accR[2] = {}, accI[2] = {};
#pragma unroll
  for (int kc = 0; kc < 4; ++kc) {
    int k0 = kc * 32;
    v16h ar = frag_a_g(hdn_rh + (rowBase + m0) * FF + k0, FF, lane);
    v16h ai = frag_a_g(hdn_ih + (rowBase + m0) * FF + k0, FF, lane);
    v16h nai = neg16(ai);
#pragma unroll
    for (int nt = 0; nt < 2; ++nt) {
      v16h br = frag_ld(Br + ((kc * 4 + n0w + nt) << 9), lane);
      v16h bi = frag_ld(Bi + ((kc * 4 + n0w + nt) << 9), lane);
      accR[nt] = WMMA_F16(ar, br, accR[nt]);
      accR[nt] = WMMA_F16(nai, bi, accR[nt]);
      accI[nt] = WMMA_F16(ar, bi, accI[nt]);
      accI[nt] = WMMA_F16(ai, br, accI[nt]);
    }
  }
  float pt = ptop_d[timg];
  const float* b2r = b2_re + expert * DIM;
  const float* b2i = b2_im + expert * DIM;
  float* out_im = out + (size_t)BT * DIM * HW;
  int nlo = lane & 15, hs = lane >> 4;
#pragma unroll
  for (int nt = 0; nt < 2; ++nt) {
    int d = (n0w + nt) * 16 + nlo;
    float br0 = b2r[d], bi0 = b2i[d];
#pragma unroll
    for (int r = 0; r < 8; ++r) {
      int pr = blockIdx.x * 64 + m0 + r + 8 * hs;    // pixel within image
      size_t p = (size_t)timg * HW + pr;
      float dr = pt * (accR[nt][r] + br0);
      float di = pt * (accI[nt][r] + bi0);
      float xr2 = xoc_r[p * DIM + d] + dr;
      float xi2 = xoc_i[p * DIM + d] + di;
      size_t oix = ((size_t)(timg * DIM + d)) * HW + pr;
      out[oix]    = x_re[oix] + xr2;
      out_im[oix] = x_im[oix] + xi2;
    }
  }
}

// ---------------------------------------------------------------------------
static inline void* wsalloc(char*& p, size_t bytes) {
  void* r = (void*)p;
  p += (bytes + 255) & ~(size_t)255;
  return r;
}

extern "C" void kernel_launch(void* const* d_in, const int* in_sizes, int n_in,
                              void* d_out, int out_size, void* d_ws, size_t ws_size,
                              hipStream_t stream) {
  const float* x_re    = (const float*)d_in[0];
  const float* x_im    = (const float*)d_in[1];
  const float* dt      = (const float*)d_in[4];
  const float* flux_re = (const float*)d_in[5];
  const float* flux_im = (const float*)d_in[6];
  const float* gamma_s = (const float*)d_in[7];
  const float* beta_s  = (const float*)d_in[8];
  const float* gamma_t = (const float*)d_in[9];
  const float* beta_t  = (const float*)d_in[10];
  const float* w_conv  = (const float*)d_in[11];
  const float* b_conv  = (const float*)d_in[12];
  const float* metric  = (const float*)d_in[13];
  const float* e_enc_re = (const float*)d_in[14];
  const float* e_enc_im = (const float*)d_in[15];
  const float* e_dec_re = (const float*)d_in[16];
  const float* e_dec_im = (const float*)d_in[17];
  const float* alpha_re = (const float*)d_in[18];
  const float* alpha_im = (const float*)d_in[19];
  const float* w_src_re = (const float*)d_in[20];
  const float* w_src_im = (const float*)d_in[21];
  const float* w_gate   = (const float*)d_in[22];
  const float* b_gate   = (const float*)d_in[23];
  const float* lam_raw  = (const float*)d_in[24];
  const float* lam_im   = (const float*)d_in[25];
  const float* w_r      = (const float*)d_in[26];
  const float* b_r      = (const float*)d_in[27];
  const float* w1_re    = (const float*)d_in[28];
  const float* w1_im    = (const float*)d_in[29];
  const float* b1_re    = (const float*)d_in[30];
  const float* b1_im    = (const float*)d_in[31];
  const float* w2_re    = (const float*)d_in[32];
  const float* w2_im    = (const float*)d_in[33];
  const float* b2_re    = (const float*)d_in[34];
  const float* b2_im    = (const float*)d_in[35];
  float* out = (float*)d_out;

  char* wp = (char*)d_ws;
  _Float16* xn_h   = (_Float16*)wsalloc(wp, (size_t)NPIX * C2 * 2);
  _Float16* wb_h   = (_Float16*)wsalloc(wp, (size_t)KCONV * C2 * 2);
  _Float16* ebr_h  = (_Float16*)wsalloc(wp, (size_t)DIM * DIM * 2);
  _Float16* ebi_h  = (_Float16*)wsalloc(wp, (size_t)DIM * DIM * 2);
  _Float16* dbr_h  = (_Float16*)wsalloc(wp, (size_t)DIM * DIM * 2);
  _Float16* dbi_h  = (_Float16*)wsalloc(wp, (size_t)DIM * DIM * 2);
  _Float16* xe_rh  = (_Float16*)wsalloc(wp, (size_t)NPIX * DIM * 2);
  _Float16* xe_ih  = (_Float16*)wsalloc(wp, (size_t)NPIX * DIM * 2);
  float*    xeig_r = (float*)wsalloc(wp, (size_t)NPIX * DIM * 4);
  float*    xeig_i = (float*)wsalloc(wp, (size_t)NPIX * DIM * 4);
  float*    xm_r   = (float*)wsalloc(wp, (size_t)BT * DIM * 4);
  float*    xm_i   = (float*)wsalloc(wp, (size_t)BT * DIM * 4);
  float*    gate_g = (float*)wsalloc(wp, (size_t)BT * DIM * 4);
  float*    src_r  = (float*)wsalloc(wp, (size_t)BT * DIM * 4);
  float*    src_i  = (float*)wsalloc(wp, (size_t)BT * DIM * 4);
  float*    A_r    = (float*)wsalloc(wp, (size_t)BT * DIM * 4);
  float*    A_i    = (float*)wsalloc(wp, (size_t)BT * DIM * 4);
  float*    opf_r  = (float*)wsalloc(wp, (size_t)BT * DIM * 4);
  float*    opf_i  = (float*)wsalloc(wp, (size_t)BT * DIM * 4);
  _Float16* u_rh   = (_Float16*)wsalloc(wp, (size_t)NPIX * DIM * 2);
  _Float16* u_ih   = (_Float16*)wsalloc(wp, (size_t)NPIX * DIM * 2);
  float*    xoc_r  = (float*)wsalloc(wp, (size_t)NPIX * DIM * 4);
  float*    xoc_i  = (float*)wsalloc(wp, (size_t)NPIX * DIM * 4);
  _Float16* xoc_rh = (_Float16*)wsalloc(wp, (size_t)NPIX * DIM * 2);
  _Float16* xoc_ih = (_Float16*)wsalloc(wp, (size_t)NPIX * DIM * 2);
  float*    feats  = (float*)wsalloc(wp, (size_t)BT * C2 * 4);
  int*      idx_d  = (int*)wsalloc(wp, (size_t)BT * 4);
  float*    ptop_d = (float*)wsalloc(wp, (size_t)BT * 4);
  _Float16* hdn_rh = (_Float16*)wsalloc(wp, (size_t)NPIX * FF * 2);
  _Float16* hdn_ih = (_Float16*)wsalloc(wp, (size_t)NPIX * FF * 2);

  (void)in_sizes; (void)n_in; (void)out_size; (void)ws_size;

  k_pack<<<576, 256, 0, stream>>>(w_conv, e_enc_re, e_enc_im, e_dec_re, e_dec_im,
                                  wb_h, ebr_h, ebi_h, dbr_h, dbi_h);
  k_ln_spatial<<<NPIX / 8, 256, 0, stream>>>(x_re, x_im, gamma_s, beta_s, xn_h);
  k_conv<<<NPIX / 64, 256, 0, stream>>>(xn_h, wb_h, b_conv, metric, x_re, x_im, xe_rh, xe_ih);
  k_eigen<<<NPIX / 128, 256, 0, stream>>>(xe_rh, xe_ih, ebr_h, ebi_h, xeig_r, xeig_i);
  k_xmean<<<BT * DIM, 256, 0, stream>>>(xeig_r, xeig_i, xm_r, xm_i);
  k_flux<<<1, 64, 0, stream>>>(xm_r, xm_i, flux_re, flux_im, alpha_re, alpha_im,
                               w_src_re, w_src_im, w_gate, b_gate, lam_raw, lam_im, dt,
                               gate_g, src_r, src_i, A_r, A_i, opf_r, opf_i);
  k_tscan<<<(HW * DIM) / 256, 256, 0, stream>>>(xeig_r, xeig_i, gate_g, src_r, src_i,
                                                A_r, A_i, opf_r, opf_i, u_rh, u_ih);
  k_dec<<<NPIX / 128, 256, 0, stream>>>(u_rh, u_ih, dbr_h, dbi_h, gamma_t, beta_t,
                                        xoc_r, xoc_i, xoc_rh, xoc_ih);
  k_feats<<<BT * C2, 256, 0, stream>>>(xoc_r, xoc_i, feats);
  k_router<<<1, 128, 0, stream>>>(feats, w_r, b_r, idx_d, ptop_d);
  k_moe1<<<dim3(HW / 64, BT), 256, 0, stream>>>(xoc_rh, xoc_ih, w1_re, w1_im, b1_re, b1_im,
                                                idx_d, hdn_rh, hdn_ih);
  k_moe2<<<dim3(HW / 64, BT), 256, 0, stream>>>(hdn_rh, hdn_ih, w2_re, w2_im, b2_re, b2_im,
                                                idx_d, ptop_d, xoc_r, xoc_i, x_re, x_im, out);
}
